// MinVQVAE_MultiQuery_12902081757258
// MI455X (gfx1250) — compile-verified, matrix operand order checked by reading, not run on hardware
//
#include <hip/hip_runtime.h>

// ---------------------------------------------------------------------------
// Problem constants
// ---------------------------------------------------------------------------
#define BATCH      4096
#define INPUT_DIM  4096
#define N_CAT      1024
#define DIM_LAT    128
#define N_QUERY    32
#define N_HID      2048
#define ZROWS      (BATCH * N_QUERY)          // 131072 latent rows
#define ZDIM       (N_QUERY * DIM_LAT)        // 4096

typedef __attribute__((ext_vector_type(16))) __bf16        v16bf;
typedef __attribute__((ext_vector_type(8)))  float         v8f;
typedef __attribute__((ext_vector_type(2)))  float         v2f;
typedef __attribute__((ext_vector_type(4)))  unsigned int  v4u;
typedef __attribute__((ext_vector_type(8)))  int           v8i;
typedef __attribute__((ext_vector_type(4)))  int           v4i;

#if defined(__has_builtin)
#if __has_builtin(__builtin_amdgcn_tensor_load_to_lds) && \
    __has_builtin(__builtin_amdgcn_s_wait_tensorcnt)
#define USE_TDM 1
#endif
#endif
#ifndef USE_TDM
#define USE_TDM 0
#endif

__device__ __forceinline__ unsigned short f2bf(float f) {
  unsigned int u = __float_as_uint(f);
  unsigned int r = u + 0x7FFFu + ((u >> 16) & 1u);   // round-to-nearest-even
  return (unsigned short)(r >> 16);
}

__device__ __forceinline__ float gelu_exact(float v) {
  return 0.5f * v * (1.0f + erff(v * 0.70710678118654752f));
}

// ---------------------------------------------------------------------------
// GEMM geometry: block tile 128x128, BK=32 (bf16), 256 threads = 8 waves (2x4),
// wave tile 64x32 = 4x2 WMMA accumulators. LDS tiles row-major bf16, row
// stride 20 DWORDs (16 data + 4 pad -> 16B aligned rows, low bank conflict).
// ---------------------------------------------------------------------------
#define BM 128
#define BN 128
#define BK 32
#define LDW 40   // LDS row stride in ushorts (20 dwords)

#if USE_TDM
// Issue a TDM 2-D tile load: 128 rows x 32 bf16 elements from a row-major
// [rows x K] bf16 tensor into LDS at lds_off, padding 4 DWORDs every 16.
__device__ __forceinline__ void tdm_load_tile(unsigned lds_off,
                                              const unsigned short* gptr, int K)
{
  unsigned long long ga = (unsigned long long)(size_t)gptr;
  // --- D# group 0: count=1 | lds_addr | global_addr | type=2
  v4u g0 = { 1u,
             lds_off,
             (unsigned)(ga & 0xFFFFFFFFu),
             (unsigned)((ga >> 32) & 0x01FFFFFFu) | (2u << 30) };
  // --- D# group 1
  // d0: wg_mask=0 | data_size=1(2B)<<16 | pad_en<<20 | pad_interval=3(16dw)<<22
  //     | pad_amount=3(4dw)<<25
  // d1: abar_addr[15:0]=0 | tensor_dim0[15:0]<<16
  // d2: tensor_dim0[31:16] | tensor_dim1[15:0]<<16
  // d3: tensor_dim1[31:16] | tile_dim0(32)<<16
  // d4: tile_dim1(128) | tile_dim2(0)<<16
  // d5: tensor_dim0_stride[31:0] ; d6: stride0[47:32] | stride1[15:0]<<16 ; d7: 0
  const unsigned dim1 = 1u << 20;     // generous; tiles never go OOB
  v8i g1 = { (int)((1u << 16) | (1u << 20) | (3u << 22) | (3u << 25)),
             (int)(((unsigned)K & 0xFFFFu) << 16),
             (int)((((unsigned)K >> 16) & 0xFFFFu) | ((dim1 & 0xFFFFu) << 16)),
             (int)(((dim1 >> 16) & 0xFFFFu) | (32u << 16)),
             (int)128,
             (int)K,
             0, 0 };
  v4i gz4 = { 0, 0, 0, 0 };
  v8i gz8 = { 0, 0, 0, 0, 0, 0, 0, 0 };
  __builtin_amdgcn_tensor_load_to_lds(g0, g1, gz4, gz4, gz8, 0);
}
#endif

__device__ __forceinline__ void tile_copy(const unsigned short* __restrict__ A,
                                          const unsigned short* __restrict__ Bt,
                                          unsigned short* dA, unsigned short* dB,
                                          int bm0, int bn0, int K, int k0,
                                          int tid, int wave)
{
#if USE_TDM
  if (wave == 0) {
    tdm_load_tile((unsigned)(size_t)(void*)dA, A  + (size_t)bm0 * K + k0, K);
    tdm_load_tile((unsigned)(size_t)(void*)dB, Bt + (size_t)bn0 * K + k0, K);
  }
#else
#pragma unroll
  for (int h = 0; h < 2; ++h) {
    const int c   = tid + h * 256;   // 512 x 16B chunks per tile
    const int row = c >> 2;
    const int off = (c & 3) * 8;     // ushorts
    *(uint4*)&dA[row * LDW + off] = *(const uint4*)(A  + (size_t)(bm0 + row) * K + k0 + off);
    *(uint4*)&dB[row * LDW + off] = *(const uint4*)(Bt + (size_t)(bn0 + row) * K + k0 + off);
  }
#endif
}

// C[M,N] = act(A[M,K](bf16) * Bt[N,K](bf16)^T + bias[N])
// ACT: 0 none, 1 exact GELU, 2 sigmoid.  OUTBF: 0 f32 out, 1 bf16 out.
template<int ACT, int OUTBF>
__global__ __launch_bounds__(256)
void gemm_bf16(const unsigned short* __restrict__ A,
               const unsigned short* __restrict__ Bt,
               const float* __restrict__ bias, void* __restrict__ Cout,
               int M, int N, int K)
{
  __shared__ __align__(16) unsigned short sA[2][BM * LDW];
  __shared__ __align__(16) unsigned short sB[2][BN * LDW];

  const int tid  = threadIdx.x;
  const int lane = tid & 31;
  const int wave = tid >> 5;
  const int wm   = wave >> 2;          // 0..1 -> 64-row slab
  const int wn   = wave & 3;           // 0..3 -> 32-col slab
  const int bm0  = blockIdx.y * BM;
  const int bn0  = blockIdx.x * BN;
  const int lm   = lane & 15;
  const int kh   = lane >> 4;

  v8f acc[4][2];
#pragma unroll
  for (int mi = 0; mi < 4; ++mi)
#pragma unroll
    for (int ni = 0; ni < 2; ++ni) {
      v8f z = {0.f,0.f,0.f,0.f,0.f,0.f,0.f,0.f};
      acc[mi][ni] = z;
    }

  const int niter = K / BK;
  tile_copy(A, Bt, sA[0], sB[0], bm0, bn0, K, 0, tid, wave);

  for (int it = 0; it < niter; ++it) {
#if USE_TDM
    if (wave == 0) __builtin_amdgcn_s_wait_tensorcnt((short)0);
#endif
    __syncthreads();   // buffer (it&1) ready; everyone done with the other one
    if (it + 1 < niter)
      tile_copy(A, Bt, sA[(it + 1) & 1], sB[(it + 1) & 1],
                bm0, bn0, K, (it + 1) * BK, tid, wave);

    const unsigned short* sAc = sA[it & 1];
    const unsigned short* sBc = sB[it & 1];

    union UF { v16bf v; unsigned int d[8]; };
    v16bf afr[4], bfr[2];
#pragma unroll
    for (int mi = 0; mi < 4; ++mi) {
      const unsigned short* base = &sAc[(wm * 64 + mi * 16 + lm) * LDW];
      UF u;
#pragma unroll
      for (int v = 0; v < 8; ++v)
        u.d[v] = *(const unsigned int*)&base[((v & 3) << 1) + (kh << 3) + ((v >> 2) << 4)];
      afr[mi] = u.v;
    }
#pragma unroll
    for (int ni = 0; ni < 2; ++ni) {
      const unsigned short* base = &sBc[(wn * 32 + ni * 16 + lm) * LDW];
      UF u;
#pragma unroll
      for (int v = 0; v < 8; ++v)
        u.d[v] = *(const unsigned int*)&base[((v & 3) << 1) + (kh << 3) + ((v >> 2) << 4)];
      bfr[ni] = u.v;
    }
#pragma unroll
    for (int mi = 0; mi < 4; ++mi)
#pragma unroll
      for (int ni = 0; ni < 2; ++ni)
        acc[mi][ni] = __builtin_amdgcn_wmma_f32_16x16x32_bf16(
            false, afr[mi], false, bfr[ni], (short)0, acc[mi][ni], false, false);
  }

  // epilogue: bias + activation
#pragma unroll
  for (int ni = 0; ni < 2; ++ni) {
    const int col = bn0 + wn * 32 + ni * 16 + lm;
    const float bv = bias[col];
#pragma unroll
    for (int mi = 0; mi < 4; ++mi) {
      const int rowbase = bm0 + wm * 64 + mi * 16 + (kh << 3);
#pragma unroll
      for (int r = 0; r < 8; ++r) {
        float v = acc[mi][ni][r] + bv;
        if (ACT == 1) v = gelu_exact(v);
        if (ACT == 2) v = 1.0f / (1.0f + __expf(-v));
        if (OUTBF) ((unsigned short*)Cout)[(size_t)(rowbase + r) * N + col] = f2bf(v);
        else       ((float*)Cout)[(size_t)(rowbase + r) * N + col] = v;
      }
    }
  }
}

// ---------------------------------------------------------------------------
// f32 -> bf16 elementwise (8 per thread)
// ---------------------------------------------------------------------------
__global__ __launch_bounds__(256)
void cvt_bf16(const float* __restrict__ in, unsigned short* __restrict__ out)
{
  size_t i = ((size_t)blockIdx.x * 256 + threadIdx.x) * 8;
  float4 a = *(const float4*)(in + i);
  float4 b = *(const float4*)(in + i + 4);
  uint4 o;
  o.x = (unsigned)f2bf(a.x) | ((unsigned)f2bf(a.y) << 16);
  o.y = (unsigned)f2bf(a.z) | ((unsigned)f2bf(a.w) << 16);
  o.z = (unsigned)f2bf(b.x) | ((unsigned)f2bf(b.y) << 16);
  o.w = (unsigned)f2bf(b.z) | ((unsigned)f2bf(b.w) << 16);
  *(uint4*)(out + i) = o;
}

// ---------------------------------------------------------------------------
// W[K,N] f32 -> Wt[N,K] bf16 (32x32 LDS tile transpose), block (32,8)
// ---------------------------------------------------------------------------
__global__ __launch_bounds__(256)
void transpose_bf16(const float* __restrict__ W, unsigned short* __restrict__ Wt,
                    int K, int N)
{
  __shared__ float t[32][33];
  const int tx = threadIdx.x, ty = threadIdx.y;
  const int k0 = blockIdx.y * 32, n0 = blockIdx.x * 32;
#pragma unroll
  for (int i = 0; i < 4; ++i)
    t[ty + i * 8][tx] = W[(size_t)(k0 + ty + i * 8) * N + n0 + tx];
  __syncthreads();
#pragma unroll
  for (int i = 0; i < 4; ++i)
    Wt[(size_t)(n0 + ty + i * 8) * K + k0 + tx] = f2bf(t[tx][ty + i * 8]);
}

// ---------------------------------------------------------------------------
// embed-pool squared norms
// ---------------------------------------------------------------------------
__global__ void enorm_kernel(const float* __restrict__ embed, float* __restrict__ enorm)
{
  int c = blockIdx.x * blockDim.x + threadIdx.x;
  if (c >= N_CAT) return;
  const float* e = embed + (size_t)c * DIM_LAT;
  float s = 0.f;
#pragma unroll 8
  for (int d = 0; d < DIM_LAT; ++d) { float v = e[d]; s += v * v; }
  enorm[c] = s;
}

// ---------------------------------------------------------------------------
// VQ argmin via fp32 WMMA (16x16x4): score = z.e - 0.5*||e||^2, argmax score.
// ---------------------------------------------------------------------------
__global__ __launch_bounds__(256)
void vq_kernel(const float* __restrict__ ze, const float* __restrict__ embed,
               const float* __restrict__ enorm, int* __restrict__ zidx,
               float* __restrict__ zdisc)
{
  __shared__ __align__(16) float sE[16 * DIM_LAT];
  __shared__ float sEn[16];
  __shared__ float red_s[8 * 256];
  __shared__ int   red_i[8 * 256];

  const int tid  = threadIdx.x;
  const int lane = tid & 31;
  const int wave = tid >> 5;
  const int lm   = lane & 15;
  const int kh   = lane >> 4;
  const int row0 = blockIdx.x * 128 + wave * 16;

  float2 areg[32];
  {
    const float* zrow = ze + (size_t)(row0 + lm) * DIM_LAT + (kh << 1);
#pragma unroll
    for (int kk = 0; kk < 32; ++kk)
      areg[kk] = *(const float2*)(zrow + kk * 4);
  }

  float best[8];
  int   bidx[8];
#pragma unroll
  for (int r = 0; r < 8; ++r) { best[r] = -3.4e38f; bidx[r] = 0; }

  for (int n0 = 0; n0 < N_CAT; n0 += 16) {
    __syncthreads();
#pragma unroll
    for (int it = 0; it < 2; ++it) {
      int q = it * 256 + tid;
      ((float4*)sE)[q] = ((const float4*)(embed + (size_t)n0 * DIM_LAT))[q];
    }
    if (tid < 16) sEn[tid] = enorm[n0 + tid];
    __syncthreads();

    const float nb = -0.5f * sEn[lm];
    v8f c = {nb, nb, nb, nb, nb, nb, nb, nb};
    const float* bbase = sE + lm * DIM_LAT + (kh << 1);
#pragma unroll
    for (int kk = 0; kk < 32; ++kk) {
      v2f a = {areg[kk].x, areg[kk].y};
      float2 bf = *(const float2*)(bbase + kk * 4);
      v2f b = {bf.x, bf.y};
      c = __builtin_amdgcn_wmma_f32_16x16x4_f32(
              false, a, false, b, (short)0, c, false, false);
    }
#pragma unroll
    for (int r = 0; r < 8; ++r) {
      float s = c[r];
      if (s > best[r]) { best[r] = s; bidx[r] = n0 + lm; }
    }
  }

  __syncthreads();
#pragma unroll
  for (int r = 0; r < 8; ++r) {
    red_s[wave * 256 + r * 32 + lane] = best[r];
    red_i[wave * 256 + r * 32 + lane] = bidx[r];
  }
  __syncthreads();
  if (lane < 16) {
    const int m = lane;
    const int g = m >> 3, r = m & 7;
    float bs = -3.4e38f; int bi = 0;
    for (int l = 0; l < 16; ++l) {
      const int L = g * 16 + l;
      float s = red_s[wave * 256 + r * 32 + L];
      int   i = red_i[wave * 256 + r * 32 + L];
      if (s > bs || (s == bs && i < bi)) { bs = s; bi = i; }
    }
    const int row = row0 + m;
    zidx[row] = bi;
    zdisc[(size_t)row * N_CAT + bi] = 1.0f;
  }
}

// ---------------------------------------------------------------------------
// gather z_q rows as bf16 (decoder input):  zqb[row][:] = bf16(embed[idx])
// ---------------------------------------------------------------------------
__global__ void gather_zq_bf16(const float* __restrict__ embed,
                               const int* __restrict__ zidx,
                               unsigned short* __restrict__ zqb)
{
  int g = blockIdx.x * blockDim.x + threadIdx.x;   // one float4 each
  int row = g >> 5;
  int j   = g & 31;
  int c = zidx[row];
  float4 v = ((const float4*)embed)[(size_t)c * 32 + j];
  uint2 p;
  p.x = (unsigned)f2bf(v.x) | ((unsigned)f2bf(v.y) << 16);
  p.y = (unsigned)f2bf(v.z) | ((unsigned)f2bf(v.w) << 16);
  *(uint2*)(zqb + (size_t)row * DIM_LAT + j * 4) = p;
}

// ---------------------------------------------------------------------------
// loss partials: dx^2*SX + dz^2*SZ, z_q gathered on the fly via zidx
// loss = (mse(x,xp) + 1.25*mse(ze,zq)) / B ; Q*D == INPUT_DIM so SZ=1.25*SX
// ---------------------------------------------------------------------------
__global__ __launch_bounds__(256)
void loss_partial(const float* __restrict__ x,  const float* __restrict__ xp,
                  const float* __restrict__ ze, const float* __restrict__ embed,
                  const int* __restrict__ zidx, float* __restrict__ partial)
{
  const float SX = 1.0f / ((float)BATCH * (float)INPUT_DIM * (float)BATCH);
  const float SZ = 1.25f * SX;
  __shared__ float sm[256];
  size_t i = ((size_t)blockIdx.x * 256 + threadIdx.x) * 4;
  float4 a = *(const float4*)(x  + i);
  float4 b = *(const float4*)(xp + i);
  float4 c = *(const float4*)(ze + i);
  const int row = (int)(i >> 7);          // 128 latent dims per row
  const int d   = (int)(i & 127);
  float4 e = *(const float4*)(embed + (size_t)zidx[row] * DIM_LAT + d);
  float s = 0.f;
  { float t = a.x - b.x; s += t * t * SX; }
  { float t = a.y - b.y; s += t * t * SX; }
  { float t = a.z - b.z; s += t * t * SX; }
  { float t = a.w - b.w; s += t * t * SX; }
  { float t = c.x - e.x; s += t * t * SZ; }
  { float t = c.y - e.y; s += t * t * SZ; }
  { float t = c.z - e.z; s += t * t * SZ; }
  { float t = c.w - e.w; s += t * t * SZ; }
  sm[threadIdx.x] = s;
  __syncthreads();
  for (int off = 128; off > 0; off >>= 1) {
    if ((int)threadIdx.x < off) sm[threadIdx.x] += sm[threadIdx.x + off];
    __syncthreads();
  }
  if (threadIdx.x == 0) partial[blockIdx.x] = sm[0];
}

__global__ __launch_bounds__(256)
void loss_final(const float* __restrict__ partial, int n, float* __restrict__ out)
{
  __shared__ float sm[256];
  float s = 0.f;
  for (int i = threadIdx.x; i < n; i += 256) s += partial[i];
  sm[threadIdx.x] = s;
  __syncthreads();
  for (int off = 128; off > 0; off >>= 1) {
    if ((int)threadIdx.x < off) sm[threadIdx.x] += sm[threadIdx.x + off];
    __syncthreads();
  }
  if (threadIdx.x == 0) *out = sm[0];
}

// ---------------------------------------------------------------------------
// launcher
// ---------------------------------------------------------------------------
extern "C" void kernel_launch(void* const* d_in, const int* in_sizes, int n_in,
                              void* d_out, int out_size, void* d_ws, size_t ws_size,
                              hipStream_t stream)
{
  (void)in_sizes; (void)n_in; (void)out_size; (void)ws_size;

  const float* x      = (const float*)d_in[0];
  const float* embed  = (const float*)d_in[1];
  const float* enc_w1 = (const float*)d_in[2];
  const float* enc_b1 = (const float*)d_in[3];
  const float* enc_w2 = (const float*)d_in[4];
  const float* enc_b2 = (const float*)d_in[5];
  const float* enc_w3 = (const float*)d_in[6];
  const float* enc_b3 = (const float*)d_in[7];
  const float* dec_w1 = (const float*)d_in[8];
  const float* dec_b1 = (const float*)d_in[9];
  const float* dec_w2 = (const float*)d_in[10];
  const float* dec_b2 = (const float*)d_in[11];
  const float* dec_w3 = (const float*)d_in[12];
  const float* dec_b3 = (const float*)d_in[13];

  // output layout: x_pred [B*4096] | z_discrete [B*32*1024] | loss [1]
  float* xpred = (float*)d_out;
  float* zdisc = (float*)d_out + (size_t)BATCH * INPUT_DIM;
  float* loss  = zdisc + (size_t)ZROWS * N_CAT;

  // workspace layout
  char* w = (char*)d_ws;
  float* ze   = (float*)w;            w += (size_t)BATCH * ZDIM * 4;
  float* enrm = (float*)w;            w += (size_t)N_CAT * 4;
  float* part = (float*)w;            w += (size_t)16384 * 4;
  int*   zidx = (int*)w;              w += (size_t)ZROWS * 4;
  unsigned short* xb   = (unsigned short*)w;  w += (size_t)BATCH * INPUT_DIM * 2;
  unsigned short* h1b  = (unsigned short*)w;  w += (size_t)BATCH * N_HID * 2;
  unsigned short* h2b  = (unsigned short*)w;  w += (size_t)BATCH * N_HID * 2;
  unsigned short* zqb  = (unsigned short*)w;  w += (size_t)BATCH * ZDIM * 2;
  unsigned short* wt1e = (unsigned short*)w;  w += (size_t)INPUT_DIM * N_HID * 2;
  unsigned short* wt2e = (unsigned short*)w;  w += (size_t)N_HID * N_HID * 2;
  unsigned short* wt3e = (unsigned short*)w;  w += (size_t)N_HID * ZDIM * 2;
  unsigned short* wt1d = (unsigned short*)w;  w += (size_t)ZDIM * N_HID * 2;
  unsigned short* wt2d = (unsigned short*)w;  w += (size_t)N_HID * N_HID * 2;
  unsigned short* wt3d = (unsigned short*)w;  w += (size_t)N_HID * INPUT_DIM * 2;

  // zero the one-hot region + loss scalar (contiguous)
  (void)hipMemsetAsync(zdisc, 0, ((size_t)ZROWS * N_CAT + 1) * sizeof(float), stream);

  dim3 blk(256);
  dim3 tblk(32, 8);

  // precision conversions (bf16 operands for the WMMA GEMMs)
  cvt_bf16<<<dim3((BATCH * INPUT_DIM) / (256 * 8)), blk, 0, stream>>>(x, xb);
  transpose_bf16<<<dim3(N_HID / 32, INPUT_DIM / 32), tblk, 0, stream>>>(enc_w1, wt1e, INPUT_DIM, N_HID);
  transpose_bf16<<<dim3(N_HID / 32, N_HID / 32),     tblk, 0, stream>>>(enc_w2, wt2e, N_HID, N_HID);
  transpose_bf16<<<dim3(ZDIM / 32, N_HID / 32),      tblk, 0, stream>>>(enc_w3, wt3e, N_HID, ZDIM);
  transpose_bf16<<<dim3(N_HID / 32, ZDIM / 32),      tblk, 0, stream>>>(dec_w1, wt1d, ZDIM, N_HID);
  transpose_bf16<<<dim3(N_HID / 32, N_HID / 32),     tblk, 0, stream>>>(dec_w2, wt2d, N_HID, N_HID);
  transpose_bf16<<<dim3(INPUT_DIM / 32, N_HID / 32), tblk, 0, stream>>>(dec_w3, wt3d, N_HID, INPUT_DIM);

  enorm_kernel<<<dim3((N_CAT + 255) / 256), blk, 0, stream>>>(embed, enrm);

  // encoder
  gemm_bf16<1, 1><<<dim3(N_HID / BN, BATCH / BM), blk, 0, stream>>>(
      xb, wt1e, enc_b1, h1b, BATCH, N_HID, INPUT_DIM);
  gemm_bf16<1, 1><<<dim3(N_HID / BN, BATCH / BM), blk, 0, stream>>>(
      h1b, wt2e, enc_b2, h2b, BATCH, N_HID, N_HID);
  gemm_bf16<0, 0><<<dim3(ZDIM / BN, BATCH / BM), blk, 0, stream>>>(
      h2b, wt3e, enc_b3, ze, BATCH, ZDIM, N_HID);

  // vector quantization (fp32 WMMA) + one-hot scatter + bf16 gather
  vq_kernel<<<dim3(ZROWS / 128), blk, 0, stream>>>(ze, embed, enrm, zidx, zdisc);
  gather_zq_bf16<<<dim3((ZROWS * 32) / 256), blk, 0, stream>>>(embed, zidx, zqb);

  // decoder (straight-through: input is z_q)
  gemm_bf16<1, 1><<<dim3(N_HID / BN, BATCH / BM), blk, 0, stream>>>(
      zqb, wt1d, dec_b1, h1b, BATCH, N_HID, ZDIM);
  gemm_bf16<1, 1><<<dim3(N_HID / BN, BATCH / BM), blk, 0, stream>>>(
      h1b, wt2d, dec_b2, h2b, BATCH, N_HID, N_HID);
  gemm_bf16<2, 0><<<dim3(INPUT_DIM / BN, BATCH / BM), blk, 0, stream>>>(
      h2b, wt3d, dec_b3, xpred, BATCH, INPUT_DIM, N_HID);

  // loss (deterministic two-stage reduction)
  const int nblocks = (BATCH * INPUT_DIM) / 1024;   // 16384
  loss_partial<<<dim3(nblocks), blk, 0, stream>>>(x, xpred, ze, embed, zidx, part);
  loss_final<<<dim3(1), blk, 0, stream>>>(part, nblocks, loss);
}
